// Attention_61177514164290
// MI455X (gfx1250) — matrix operand order repaired and checked by reading, not compiled, not run on hardware
//
#include <hip/hip_runtime.h>

// ---------------------------------------------------------------------------
// Gated RoPE attention for MI455X (gfx1250, wave32, WMMA f32_16x16x32_f16)
// - contraction-contiguous operands -> b128 loads
// - software-pipelined GEMM k-loops (staggered loadcnt waits under WMMAs)
// - attention: block-shared K/V staged in LDS via global_load_async_to_lds_b128
//   (double-buffered, s_wait_asynccnt), 8x less L2 traffic per block
// ---------------------------------------------------------------------------

typedef _Float16 half_t;
typedef __attribute__((ext_vector_type(16))) _Float16 v16h;
typedef __attribute__((ext_vector_type(8)))  float    v8f;

#define B_    2
#define N_    4096
#define D_    512
#define H_    8
#define DH_   64
#define QKV_  1536
#define ROWS_ (B_ * N_)      // 8192

// ---------------------------------------------------------------------------
// Fragment loaders (wave32), contiguous k order k = 16*(lane/16) + e in A and B
// ---------------------------------------------------------------------------

// A-matrix 16x32 f16, row-major source with leading dim `ld`.
static __device__ __forceinline__ v16h load_a_frag(const half_t* src, int ld) {
  const int l  = threadIdx.x & 31;
  const half_t* row = src + (l & 15) * ld + (l >> 4) * 16;
  v16h a;
#pragma unroll
  for (int e = 0; e < 16; ++e) {
    const int kk = e;   // one contiguous 16-elem run
    a[e] = row[kk];
  }
  return a;
}

// B-matrix 32x16 f16 with contiguous contraction dim:
// B(k, n) = src[n * row_stride + k] -> 2x b128 per lane (global or LDS).
static __device__ __forceinline__ v16h load_bT_contig(const half_t* src,
                                                      int row_stride) {
  const int l = threadIdx.x & 31;
  const half_t* p = src + (l & 15) * row_stride + (l >> 4) * 16;
  return *(const v16h*)p;   // 32-byte aligned for all our buffers
}

static __device__ __forceinline__ v8f wmma_f16(v16h a, v16h b, v8f c) {
  return __builtin_amdgcn_wmma_f32_16x16x32_f16(false, a, false, b, (short)0, c,
                                                false, false);
}

// Async 16-byte global -> LDS copy (CDNA5 VGLOBAL async encoding, ASYNCcnt).
static __device__ __forceinline__ void async_ld16(void* lds_dst,
                                                  const void* gsrc) {
  const unsigned loff = (unsigned)(uintptr_t)lds_dst;   // LDS byte offset
  const unsigned long long ga = (unsigned long long)(uintptr_t)gsrc;
  asm volatile("global_load_async_to_lds_b128 %0, %1, off"
               :: "v"(loff), "v"(ga)
               : "memory");
}

static __device__ __forceinline__ void wait_async0() {
  asm volatile("s_wait_asynccnt 0x0" ::: "memory");
}

// ---------------------------------------------------------------------------
// 1a) fp32 -> f16 (x)
// ---------------------------------------------------------------------------
__global__ void cvt_f32_to_f16(const float* __restrict__ in,
                               half_t* __restrict__ out, int n) {
  int i = blockIdx.x * blockDim.x + threadIdx.x;
  if (i < n) out[i] = (half_t)in[i];
}

// 1b) fp32 [R,C] -> f16 transposed [C,R]
__global__ void cvt_transpose_f16(const float* __restrict__ in,
                                  half_t* __restrict__ out, int R, int C) {
  int idx = blockIdx.x * blockDim.x + threadIdx.x;
  if (idx >= R * C) return;
  const int r = idx / C, c = idx % C;
  out[(size_t)c * R + r] = (half_t)in[idx];
}

// ---------------------------------------------------------------------------
// 2) QKV projection: C[8192,1536] = Xf16[8192,512] @ W[512,1536]
//    W pre-transposed [1536,512]. Wave tile 32x64, software-pipelined k-loop.
//    q,k -> [b*h, n, 64]; v -> TRANSPOSED [b*h, 64, n].
// ---------------------------------------------------------------------------
__global__ void qkv_gemm_kernel(const half_t* __restrict__ xh,
                                const half_t* __restrict__ wt,
                                half_t* __restrict__ qb,
                                half_t* __restrict__ kb,
                                half_t* __restrict__ vt) {
  const int wave = (blockIdx.x * blockDim.x + threadIdx.x) >> 5;
  const int tr   = wave / (QKV_ / 64);          // 256 row tiles of 32
  const int tcs  = wave % (QKV_ / 64);          // 24 col strips of 64
  const int row0 = tr * 32, col0 = tcs * 64;

  v8f acc[2][4] = {};
  v16h a[2], b[4];
#pragma unroll
  for (int s = 0; s < 2; ++s)
    a[s] = load_a_frag(xh + (size_t)(row0 + s * 16) * D_, D_);
#pragma unroll
  for (int f = 0; f < 4; ++f)
    b[f] = load_bT_contig(wt + (size_t)(col0 + f * 16) * D_, D_);

  for (int k0 = 0; k0 < D_; k0 += 32) {
    const int kn = (k0 + 32) & (D_ - 1);        // wrap: last prefetch is benign
    v16h an[2], bn[4];
#pragma unroll
    for (int s = 0; s < 2; ++s)
      an[s] = load_a_frag(xh + (size_t)(row0 + s * 16) * D_ + kn, D_);
#pragma unroll
    for (int f = 0; f < 4; ++f)
      bn[f] = load_bT_contig(wt + (size_t)(col0 + f * 16) * D_ + kn, D_);
#pragma unroll
    for (int s = 0; s < 2; ++s)
#pragma unroll
      for (int f = 0; f < 4; ++f)
        acc[s][f] = wmma_f16(a[s], b[f], acc[s][f]);
#pragma unroll
    for (int s = 0; s < 2; ++s) a[s] = an[s];
#pragma unroll
    for (int f = 0; f < 4; ++f) b[f] = bn[f];
  }

  const int l  = threadIdx.x & 31;
  const int nc = l & 15, hg = l >> 4;
  const int sect = col0 >> 9;                   // 0=q 1=k 2=v
  const int head = (col0 & 511) >> 6;

  if (sect < 2) {
    half_t* dst = (sect == 0) ? qb : kb;
#pragma unroll
    for (int s = 0; s < 2; ++s)
#pragma unroll
      for (int r = 0; r < 8; ++r) {
        const int grow = row0 + s * 16 + hg * 8 + r;
        const int bi = grow >> 12, t = grow & (N_ - 1);
#pragma unroll
        for (int f = 0; f < 4; ++f)
          dst[(((size_t)(bi * H_ + head) * N_) + t) * DH_ + f * 16 + nc] =
              (half_t)acc[s][f][r];
      }
  } else {
#pragma unroll
    for (int s = 0; s < 2; ++s)
#pragma unroll
      for (int r = 0; r < 8; ++r) {
        const int grow = row0 + s * 16 + hg * 8 + r;
        const int bi = grow >> 12, t = grow & (N_ - 1);
#pragma unroll
        for (int f = 0; f < 4; ++f)
          vt[((size_t)(bi * H_ + head) * DH_ + f * 16 + nc) * N_ + t] =
              (half_t)acc[s][f][r];
      }
  }
}

// ---------------------------------------------------------------------------
// 3) In-place RoPE on a [b*h, n, 64] f16 buffer (one thread per rotation pair)
// ---------------------------------------------------------------------------
__global__ void rope_kernel(half_t* __restrict__ buf) {
  const int idx = blockIdx.x * blockDim.x + threadIdx.x;   // B*H*N*32 pairs
  const int p  = idx & 31;
  const int t  = (idx >> 5) & (N_ - 1);
  const int bh = idx >> 17;
  if (bh >= B_ * H_) return;
  const int j = p & 15;                         // concat([freqs,freqs])
  const float freq = (float)t * __powf(10.0f, -(float)j * (1.0f / 16.0f));
  float s, c;
  __sincosf(freq, &s, &c);
  half_t* el = buf + ((size_t)(bh * N_ + t) * DH_) + 2 * p;
  const float x1 = (float)el[0], x2 = (float)el[1];
  el[0] = (half_t)(x1 * c - x2 * s);
  el[1] = (half_t)(x1 * s + x2 * c);
}

// ---------------------------------------------------------------------------
// 4) Per-head sigmoid gates: gates[b*n, 8] = sigmoid(x @ w_gates + b_gates)
// ---------------------------------------------------------------------------
__global__ void gates_kernel(const float* __restrict__ x,
                             const float* __restrict__ wg,
                             const float* __restrict__ bg,
                             float* __restrict__ gates) {
  __shared__ float xs[D_];
  __shared__ float partial[128];
  const int row = blockIdx.x;                   // 8192 rows
  const float* xr = x + (size_t)row * D_;
  for (int i = threadIdx.x; i < D_; i += blockDim.x) xs[i] = xr[i];
  __syncthreads();

  const int h     = threadIdx.x & 7;
  const int chunk = threadIdx.x >> 3;           // 16 chunks of 32
  float sum = 0.f;
  const int k0 = chunk * 32;
#pragma unroll 8
  for (int k = k0; k < k0 + 32; ++k) sum += xs[k] * wg[k * H_ + h];
  partial[threadIdx.x] = sum;
  __syncthreads();

  if (threadIdx.x < H_) {
    float s = bg[threadIdx.x];
    for (int c = 0; c < 16; ++c) s += partial[c * H_ + threadIdx.x];
    gates[row * H_ + threadIdx.x] = 1.0f / (1.0f + __expf(-s));
  }
}

// ---------------------------------------------------------------------------
// 5) Flash attention. Block = 8 waves on the SAME (b,h), 8 consecutive 32-row
//    Q tiles. K/V tiles (shared by all 8 waves) are async-copied into LDS,
//    double-buffered: copy of tile i+1 overlaps compute of tile i.
// ---------------------------------------------------------------------------
__global__ void attn_kernel(const half_t* __restrict__ qb,
                            const half_t* __restrict__ kb,
                            const half_t* __restrict__ vt,   // [b*h, 64, n]
                            const float*  __restrict__ gates,
                            half_t* __restrict__ attnout) {
  __shared__ half_t kbuf[2][32 * DH_];          // 2 x 4 KB  (rows kv, cols dh)
  __shared__ half_t vbuf[2][DH_ * 32];          // 2 x 4 KB  (rows dh, cols kv)
  __shared__ half_t plds[8][2][16 * 32];        // per-wave P staging, 16 KB

  const int tid  = threadIdx.x;
  const int wib  = tid >> 5;
  const int l    = tid & 31;
  const int nc   = l & 15, hg = l >> 4;
  const int task = blockIdx.x * 8 + wib;        // 2048 tasks
  const int bh    = task >> 7;                  // all 8 waves share bh
  const int qtile = task & 127;
  const int q0    = qtile * 32;

  const half_t* Q  = qb + ((size_t)(bh * N_ + q0)) * DH_;
  const half_t* K  = kb + (size_t)bh * N_ * DH_;
  const half_t* Vt = vt + (size_t)bh * DH_ * N_;

  // cooperative async tile copy: 256 threads x 16B each for K and V
  auto stage_tiles = [&](int buf, int kv0) {
    async_ld16(&kbuf[buf][tid * 8], K + (size_t)kv0 * DH_ + tid * 8);
    const int seg = tid >> 2, qq = tid & 3;     // 64 dh-rows x 4 chunks
    async_ld16(&vbuf[buf][seg * 32 + qq * 8],
               Vt + (size_t)seg * N_ + kv0 + qq * 8);
  };

  v16h qf[2][2];
#pragma unroll
  for (int s = 0; s < 2; ++s) {
    qf[s][0] = load_a_frag(Q + s * 16 * DH_ + 0,  DH_);
    qf[s][1] = load_a_frag(Q + s * 16 * DH_ + 32, DH_);
  }

  float m_i[2][8], l_i[2][8];
#pragma unroll
  for (int s = 0; s < 2; ++s)
#pragma unroll
    for (int r = 0; r < 8; ++r) { m_i[s][r] = -1.0e30f; l_i[s][r] = 0.0f; }
  v8f o[2][4] = {};                             // 2 x (16 x 64) f32

  const float scale = 0.125f;                   // 64^-0.5

  stage_tiles(0, 0);                            // prologue copy of tile 0

  for (int it = 0; it < N_ / 32; ++it) {
    const int cur = it & 1;
    wait_async0();                              // our copies into `cur` landed
    __syncthreads();                            // all waves' copies landed;
                                                // alt buffer free to overwrite
    const int kvn = ((it + 1) & (N_ / 32 - 1)) * 32;
    stage_tiles(1 - cur, kvn);                  // overlap next copy w/ compute

    // ---- fragments from LDS (contiguous 2x ds_load_b128 per lane) ----
    v16h bk[2][2];
#pragma unroll
    for (int g = 0; g < 2; ++g) {
      bk[g][0] = load_bT_contig(&kbuf[cur][g * 16 * DH_ + 0],  DH_);
      bk[g][1] = load_bT_contig(&kbuf[cur][g * 16 * DH_ + 32], DH_);
    }
    v16h vf[4];
#pragma unroll
    for (int f = 0; f < 4; ++f)
      vf[f] = load_bT_contig(&vbuf[cur][f * 16 * 32], 32);

#pragma unroll
    for (int s = 0; s < 2; ++s) {
      v8f s0 = {}, s1 = {};
      s0 = wmma_f16(qf[s][0], bk[0][0], s0);
      s0 = wmma_f16(qf[s][1], bk[0][1], s0);
      s1 = wmma_f16(qf[s][0], bk[1][0], s1);
      s1 = wmma_f16(qf[s][1], bk[1][1], s1);

      // online softmax (rows live across 16-lane half-groups)
#pragma unroll
      for (int r = 0; r < 8; ++r) {
        s0[r] *= scale; s1[r] *= scale;
        float mv = fmaxf(s0[r], s1[r]);
        mv = fmaxf(mv, __shfl_xor(mv, 1, 32));
        mv = fmaxf(mv, __shfl_xor(mv, 2, 32));
        mv = fmaxf(mv, __shfl_xor(mv, 4, 32));
        mv = fmaxf(mv, __shfl_xor(mv, 8, 32));
        const float mnew = fmaxf(m_i[s][r], mv);
        const float alpha = __expf(m_i[s][r] - mnew);
        s0[r] = __expf(s0[r] - mnew);
        s1[r] = __expf(s1[r] - mnew);
        float rs = s0[r] + s1[r];
        rs += __shfl_xor(rs, 1, 32);
        rs += __shfl_xor(rs, 2, 32);
        rs += __shfl_xor(rs, 4, 32);
        rs += __shfl_xor(rs, 8, 32);
        l_i[s][r] = l_i[s][r] * alpha + rs;
        m_i[s][r] = mnew;
#pragma unroll
        for (int f = 0; f < 4; ++f) o[s][f][r] *= alpha;
      }

      // relayout P (C-frag -> A-frag) through wave-private LDS
      half_t* pl = &plds[wib][s][0];
#pragma unroll
      for (int r = 0; r < 8; ++r) {
        pl[(hg * 8 + r) * 32 + nc]      = (half_t)s0[r];
        pl[(hg * 8 + r) * 32 + 16 + nc] = (half_t)s1[r];
      }
    }

    __asm__ volatile("s_wait_dscnt 0" ::: "memory");

#pragma unroll
    for (int s = 0; s < 2; ++s) {
      v16h pa = load_a_frag(&plds[wib][s][0], 32);
#pragma unroll
      for (int f = 0; f < 4; ++f) o[s][f] = wmma_f16(pa, vf[f], o[s][f]);
    }
    __syncthreads();                            // done reading cur before it
                                                // becomes next iter's copy dst
  }

  // ---- epilogue: normalize, apply sigmoid gate, store f16 [b*n, 512] ----
  const int head = bh & 7, bi = bh >> 3;
#pragma unroll
  for (int s = 0; s < 2; ++s) {
#pragma unroll
    for (int r = 0; r < 8; ++r) {
      const int tok = q0 + s * 16 + hg * 8 + r;
      const float g = gates[((size_t)(bi * N_ + tok)) * H_ + head];
      const float inv = g / l_i[s][r];
#pragma unroll
      for (int f = 0; f < 4; ++f) {
        attnout[((size_t)(bi * N_ + tok)) * D_ + head * DH_ + f * 16 + nc] =
            (half_t)(o[s][f][r] * inv);
      }
    }
  }
}

// ---------------------------------------------------------------------------
// 6) Output projection: out[8192,512] = attn_f16 @ w_out + b_out (fp32 out).
// ---------------------------------------------------------------------------
__global__ void out_gemm_kernel(const half_t* __restrict__ ah,
                                const half_t* __restrict__ wt,
                                const float* __restrict__ bias,
                                float* __restrict__ out) {
  const int wave = (blockIdx.x * blockDim.x + threadIdx.x) >> 5;
  const int tr  = wave >> 3;                    // 256 row tiles of 32
  const int tcs = wave & 7;                     // 8 col strips of 64
  const int row0 = tr * 32, col0 = tcs * 64;

  v8f acc[2][4] = {};
  v16h a[2], b[4];
#pragma unroll
  for (int s = 0; s < 2; ++s)
    a[s] = load_a_frag(ah + (size_t)(row0 + s * 16) * D_, D_);
#pragma unroll
  for (int f = 0; f < 4; ++f)
    b[f] = load_bT_contig(wt + (size_t)(col0 + f * 16) * D_, D_);

  for (int k0 = 0; k0 < D_; k0 += 32) {
    const int kn = (k0 + 32) & (D_ - 1);
    v16h an[2], bn[4];
#pragma unroll
    for (int s = 0; s < 2; ++s)
      an[s] = load_a_frag(ah + (size_t)(row0 + s * 16) * D_ + kn, D_);
#pragma unroll
    for (int f = 0; f < 4; ++f)
      bn[f] = load_bT_contig(wt + (size_t)(col0 + f * 16) * D_ + kn, D_);
#pragma unroll
    for (int s = 0; s < 2; ++s)
#pragma unroll
      for (int f = 0; f < 4; ++f)
        acc[s][f] = wmma_f16(a[s], b[f], acc[s][f]);
#pragma unroll
    for (int s = 0; s < 2; ++s) a[s] = an[s];
#pragma unroll
    for (int f = 0; f < 4; ++f) b[f] = bn[f];
  }

  const int l = threadIdx.x & 31;
  const int nc = l & 15, hg = l >> 4;
#pragma unroll
  for (int s = 0; s < 2; ++s)
#pragma unroll
    for (int r = 0; r < 8; ++r) {
      const size_t rowoff = (size_t)(row0 + s * 16 + hg * 8 + r) * D_;
#pragma unroll
      for (int f = 0; f < 4; ++f)
        out[rowoff + col0 + f * 16 + nc] = acc[s][f][r] + bias[col0 + f * 16 + nc];
    }
}

// ---------------------------------------------------------------------------
// Launcher
// ---------------------------------------------------------------------------
extern "C" void kernel_launch(void* const* d_in, const int* in_sizes, int n_in,
                              void* d_out, int out_size, void* d_ws, size_t ws_size,
                              hipStream_t stream) {
  const float* x       = (const float*)d_in[0];   // [2,4096,512]
  const float* w_qkv   = (const float*)d_in[1];   // [512,1536]
  const float* w_gates = (const float*)d_in[2];   // [512,8]
  const float* b_gates = (const float*)d_in[3];   // [8]
  const float* w_out   = (const float*)d_in[4];   // [512,512]
  const float* b_out   = (const float*)d_in[5];   // [512]
  float* out = (float*)d_out;

  size_t off = 0;
  auto alloc = [&](size_t bytes) -> void* {
    void* p = (char*)d_ws + off;
    off = (off + bytes + 255) & ~(size_t)255;
    return p;
  };
  half_t* xh    = (half_t*)alloc((size_t)ROWS_ * D_ * 2);
  half_t* wqkvt = (half_t*)alloc((size_t)D_ * QKV_ * 2);   // [1536,512]
  half_t* woutt = (half_t*)alloc((size_t)D_ * D_ * 2);     // [512,512] T
  half_t* qb    = (half_t*)alloc((size_t)B_ * H_ * N_ * DH_ * 2);
  half_t* kb    = (half_t*)alloc((size_t)B_ * H_ * N_ * DH_ * 2);
  half_t* vt    = (half_t*)alloc((size_t)B_ * H_ * N_ * DH_ * 2);  // [bh,64,n]
  float*  gts   = (float*)alloc((size_t)ROWS_ * H_ * 4);
  half_t* atth  = (half_t*)alloc((size_t)ROWS_ * D_ * 2);
  (void)ws_size; (void)n_in; (void)in_sizes; (void)out_size;

  // 1) conversions (+ weight transposes)
  {
    int n = ROWS_ * D_;
    cvt_f32_to_f16<<<(n + 255) / 256, 256, 0, stream>>>(x, xh, n);
    n = D_ * QKV_;
    cvt_transpose_f16<<<(n + 255) / 256, 256, 0, stream>>>(w_qkv, wqkvt, D_, QKV_);
    n = D_ * D_;
    cvt_transpose_f16<<<(n + 255) / 256, 256, 0, stream>>>(w_out, woutt, D_, D_);
  }

  // 4) gates (independent, overlapped in stream order)
  gates_kernel<<<ROWS_, 128, 0, stream>>>(x, w_gates, b_gates, gts);

  // 2) QKV GEMM: 256 row-tiles(32) x 24 col-strips(64), 8 waves/block
  qkv_gemm_kernel<<<(256 * 24) / 8, 256, 0, stream>>>(xh, wqkvt, qb, kb, vt);

  // 3) RoPE on q and k
  {
    const int pairs = B_ * H_ * N_ * 32;
    rope_kernel<<<pairs / 256, 256, 0, stream>>>(qb);
    rope_kernel<<<pairs / 256, 256, 0, stream>>>(kb);
  }

  // 5) flash attention: 256 blocks x 8 waves (one bh stream per block)
  attn_kernel<<<2048 / 8, 256, 0, stream>>>(qb, kb, vt, gts, atth);

  // 6) output projection: 256 row-tiles(32) x 8 col-strips(64)
  out_gemm_kernel<<<(256 * 8) / 8, 256, 0, stream>>>(atth, woutt, b_out, out);
}
